// LoftQ_40355512714089
// MI455X (gfx1250) — compile-verified, matrix operand order checked by reading, not run on hardware
//
#include <hip/hip_runtime.h>
#include <hip/hip_bf16.h>

typedef __attribute__((ext_vector_type(16))) __bf16 v16bf;
typedef __attribute__((ext_vector_type(8)))  float  v8f;
typedef __attribute__((address_space(3))) unsigned short lds_ushort;

#define K_DIM   4096
#define N_DIM   4096
#define RANK    16
#define SCALING 0.25f   // 16^-0.5

#define BM 256          // block tile M (async kernel)
#define BN 128          // block tile N (async kernel)
#define BK 64
#define LDSPITCH 72     // 64 + 8 pad bf16 -> 144B rows, 16B aligned

__device__ __forceinline__ unsigned lds_addr32(const void* p) {
    return (unsigned)(unsigned long long)(const lds_ushort*)p;   // as(3) offset
}

// RNE round two f32 to bf16 and pack via one v_perm_b32
__device__ __forceinline__ unsigned pack_bf16_rne(float lo, float hi) {
    unsigned a = __float_as_uint(lo), b = __float_as_uint(hi);
    a += 0x7FFFu + ((a >> 16) & 1u);
    b += 0x7FFFu + ((b >> 16) & 1u);
    return __builtin_amdgcn_perm(b, a, 0x07060302u);   // {b_hi, a_hi}
}

// ---------------------------------------------------------------------------
// Kernel 1: W_bf16[o][d] = (W_int - zp)*scale + 0.25 * sum_r B[o][r]*A[r][d]
// ---------------------------------------------------------------------------
__global__ __launch_bounds__(256)
void loftq_prep_w(const int* __restrict__ Wint,
                  const float* __restrict__ A,
                  const float* __restrict__ Bm,
                  const float* __restrict__ scale_p,
                  const float* __restrict__ zp_p,
                  unsigned short* __restrict__ Wb) {
    const float scale = scale_p[0];
    const float zp    = zp_p[0];
    const unsigned g  = blockIdx.x * 256u + threadIdx.x;   // one thread per 8 elems
    const unsigned o  = g >> 9;                            // K_DIM/8 = 512 chunks/row
    const unsigned d0 = (g & 511u) << 3;

    float brow[RANK];
#pragma unroll
    for (int r = 0; r < RANK; ++r) brow[r] = Bm[o * RANK + r];

    float val[8];
#pragma unroll
    for (int j = 0; j < 8; ++j) {
        const unsigned d = d0 + j;
        float acc = 0.f;
#pragma unroll
        for (int r = 0; r < RANK; ++r)
            acc = fmaf(brow[r], A[r * (unsigned)K_DIM + d], acc);
        val[j] = ((float)Wint[o * (unsigned)K_DIM + d] - zp) * scale + SCALING * acc;
    }
    uint4 pk;
    pk.x = pack_bf16_rne(val[0], val[1]);
    pk.y = pack_bf16_rne(val[2], val[3]);
    pk.z = pack_bf16_rne(val[4], val[5]);
    pk.w = pack_bf16_rne(val[6], val[7]);
    *(uint4*)(Wb + (size_t)o * K_DIM + d0) = pk;
}

// ---------------------------------------------------------------------------
// Kernel 2: x fp32 -> bf16 (one streaming pass)
// ---------------------------------------------------------------------------
__global__ __launch_bounds__(256)
void loftq_prep_x(const float* __restrict__ X, unsigned short* __restrict__ Xb) {
    const size_t g = (size_t)blockIdx.x * 256u + threadIdx.x;  // 8 elems/thread
    const float4 f0 = *(const float4*)(X + g * 8);
    const float4 f1 = *(const float4*)(X + g * 8 + 4);
    uint4 pk;
    pk.x = pack_bf16_rne(f0.x, f0.y);
    pk.y = pack_bf16_rne(f0.z, f0.w);
    pk.z = pack_bf16_rne(f1.x, f1.y);
    pk.w = pack_bf16_rne(f1.z, f1.w);
    *(uint4*)(Xb + g * 8) = pk;
}

// ---------------------------------------------------------------------------
// Kernel 3 (fast path): Y = Xb * Wb^T, bf16 WMMA, f32 acc.
// 256x128 block, 64x64 wave tiles (4x4 WMMA -> 2 WMMA per LDS fragment).
// Tiles move global->LDS via GLOBAL_LOAD_ASYNC_TO_LDS_B128 (no VGPR staging),
// double-buffered LDS, one barrier per K-step.
// ---------------------------------------------------------------------------
__global__ __launch_bounds__(256, 1)
void loftq_gemm_async(const unsigned short* __restrict__ Xb,
                      const unsigned short* __restrict__ Wb,
                      float* __restrict__ Y) {
    __shared__ unsigned short ldsA[2][BM * LDSPITCH];   // 36 KB x2
    __shared__ unsigned short ldsB[2][BN * LDSPITCH];   // 18 KB x2

    const int tid  = threadIdx.x;
    const int lane = tid & 31;          // wave32
    const int wave = tid >> 5;          // 8 waves
    const int half = lane >> 4;         // K-half selector (ISA 16-bit A/B layout)
    const int mrow = lane & 15;

    const int blockM = blockIdx.y * BM;
    const int blockN = blockIdx.x * BN;
    const int wm = (wave & 3) * 64;     // wave tile 64(M) x 64(N), waves 4x2
    const int wn = (wave >> 2) * 64;

    v8f acc[4][4];
#pragma unroll
    for (int i = 0; i < 4; ++i)
#pragma unroll
        for (int j = 0; j < 4; ++j) acc[i][j] = (v8f){};

    // Per-thread async-copy slices: 8 x 16B for A (256x64), 4 x 16B for B (128x64).
    // Slice i handles row (i*32 + tid/8), 16B chunk (tid%8) of the 64-elem row.
    const int srow = tid >> 3, scol = (tid & 7) * 8;
    const unsigned short* gA0 = Xb + (size_t)(blockM + srow) * K_DIM + scol;
    const unsigned short* gB0 = Wb + (size_t)(blockN + srow) * K_DIM + scol;
    const unsigned lOff0 = (unsigned)(srow * LDSPITCH + scol) * 2u;   // bytes

    const unsigned ldsA0 = lds_addr32(&ldsA[0][0]);
    const unsigned ldsA1 = lds_addr32(&ldsA[1][0]);
    const unsigned ldsB0 = lds_addr32(&ldsB[0][0]);
    const unsigned ldsB1 = lds_addr32(&ldsB[1][0]);

    auto issueAsync = [&](int kt, int buf) {
        const unsigned aBase = (buf ? ldsA1 : ldsA0) + lOff0;
        const unsigned bBase = (buf ? ldsB1 : ldsB0) + lOff0;
        const int ke = kt * BK;                       // element offset along K
#pragma unroll
        for (int i = 0; i < 8; ++i) {                 // A: 8 slices of 32 rows
            asm volatile("global_load_async_to_lds_b128 %0, %1, off"
                         :: "v"(aBase + (unsigned)(i * 32 * LDSPITCH * 2)),
                            "v"(gA0 + (size_t)(i * 32) * K_DIM + ke) : "memory");
        }
#pragma unroll
        for (int i = 0; i < 4; ++i) {                 // B: 4 slices of 32 rows
            asm volatile("global_load_async_to_lds_b128 %0, %1, off"
                         :: "v"(bBase + (unsigned)(i * 32 * LDSPITCH * 2)),
                            "v"(gB0 + (size_t)(i * 32) * K_DIM + ke) : "memory");
        }
    };

    union Frag { v16bf v; uint4 q[2]; };

    auto computeTile = [&](int buf) {
#pragma unroll
        for (int ks = 0; ks < 2; ++ks) {              // two 32-deep WMMA substeps
            const int ko = ks * 32 + half * 8;        // lane half: K 0-7/16-23 vs 8-15/24-31
            Frag a[4], b[4];
#pragma unroll
            for (int mt = 0; mt < 4; ++mt) {
                const unsigned short* p = &ldsA[buf][(wm + mt * 16 + mrow) * LDSPITCH + ko];
                a[mt].q[0] = *(const uint4*)p;
                a[mt].q[1] = *(const uint4*)(p + 16);
            }
#pragma unroll
            for (int nt = 0; nt < 4; ++nt) {
                const unsigned short* p = &ldsB[buf][(wn + nt * 16 + mrow) * LDSPITCH + ko];
                b[nt].q[0] = *(const uint4*)p;
                b[nt].q[1] = *(const uint4*)(p + 16);
            }
#pragma unroll
            for (int mt = 0; mt < 4; ++mt)
#pragma unroll
                for (int nt = 0; nt < 4; ++nt)
                    acc[mt][nt] = __builtin_amdgcn_wmma_f32_16x16x32_bf16(
                        false, a[mt].v, false, b[nt].v,
                        (short)0, acc[mt][nt], false, false);
        }
    };

    const int ksteps = K_DIM / BK;                    // 64
    issueAsync(0, 0);
    asm volatile("s_wait_asynccnt 0x0" ::: "memory");
    __syncthreads();

    for (int kt = 0; kt < ksteps; ++kt) {
        const int buf = kt & 1;
        if (kt + 1 < ksteps) issueAsync(kt + 1, buf ^ 1);   // fill other buffer
        computeTile(buf);
        if (kt + 1 < ksteps)
            asm volatile("s_wait_asynccnt 0x0" ::: "memory");
        __syncthreads();                               // readers done + new tile visible
    }

    // Epilogue: C layout — VGPR v holds (M = v + 8*half, N = mrow)
#pragma unroll
    for (int mt = 0; mt < 4; ++mt)
#pragma unroll
        for (int nt = 0; nt < 4; ++nt)
#pragma unroll
            for (int v = 0; v < 8; ++v) {
                const int row = blockM + wm + mt * 16 + half * 8 + v;
                const int col = blockN + wn + nt * 16 + mrow;
                Y[(size_t)row * N_DIM + col] = acc[mt][nt][v];
            }
}

// ---------------------------------------------------------------------------
// Kernel 3b (fallback, ws too small): 128x128 block, convert x inline;
// nothing held across compute, single LDS buffer, two barriers per step.
// ---------------------------------------------------------------------------
__global__ __launch_bounds__(256, 1)
void loftq_gemm_f32x(const float* __restrict__ X,
                     const unsigned short* __restrict__ Wb,
                     float* __restrict__ Y) {
    __shared__ unsigned short ldsA[128 * LDSPITCH];
    __shared__ unsigned short ldsB[128 * LDSPITCH];

    const int tid  = threadIdx.x;
    const int lane = tid & 31;
    const int wave = tid >> 5;
    const int half = lane >> 4;
    const int mrow = lane & 15;

    const int blockM = blockIdx.y * 128;
    const int blockN = blockIdx.x * 128;
    const int wm = (wave & 3) * 32;
    const int wn = (wave >> 2) * 64;

    v8f acc[2][4];
#pragma unroll
    for (int i = 0; i < 2; ++i)
#pragma unroll
        for (int j = 0; j < 4; ++j) acc[i][j] = (v8f){};

    union Frag { v16bf v; uint4 q[2]; };
    const int ksteps = K_DIM / BK;

    for (int kt = 0; kt < ksteps; ++kt) {
        const int k = kt * BK;
#pragma unroll
        for (int i = 0; i < 8; ++i) {                 // A: 128x64 fp32 -> bf16
            const int linear = i * 256 + tid;
            const int row = linear >> 4, colv = linear & 15;
            const float4 f = *(const float4*)(X + (size_t)(blockM + row) * K_DIM + k + colv * 4);
            uint2 u;
            u.x = pack_bf16_rne(f.x, f.y);
            u.y = pack_bf16_rne(f.z, f.w);
            *(uint2*)&ldsA[row * LDSPITCH + colv * 4] = u;
        }
#pragma unroll
        for (int i = 0; i < 4; ++i) {                 // B: 128x64 bf16
            const int linear = i * 256 + tid;
            const int row = linear >> 3, colv = linear & 7;
            *(uint4*)&ldsB[row * LDSPITCH + colv * 8] =
                *(const uint4*)(Wb + (size_t)(blockN + row) * K_DIM + k + colv * 8);
        }
        __syncthreads();
#pragma unroll
        for (int ks = 0; ks < 2; ++ks) {
            const int ko = ks * 32 + half * 8;
            Frag a[2], b[4];
#pragma unroll
            for (int mt = 0; mt < 2; ++mt) {
                const unsigned short* p = &ldsA[(wm + mt * 16 + mrow) * LDSPITCH + ko];
                a[mt].q[0] = *(const uint4*)p;
                a[mt].q[1] = *(const uint4*)(p + 16);
            }
#pragma unroll
            for (int nt = 0; nt < 4; ++nt) {
                const unsigned short* p = &ldsB[(wn + nt * 16 + mrow) * LDSPITCH + ko];
                b[nt].q[0] = *(const uint4*)p;
                b[nt].q[1] = *(const uint4*)(p + 16);
            }
#pragma unroll
            for (int mt = 0; mt < 2; ++mt)
#pragma unroll
                for (int nt = 0; nt < 4; ++nt)
                    acc[mt][nt] = __builtin_amdgcn_wmma_f32_16x16x32_bf16(
                        false, a[mt].v, false, b[nt].v,
                        (short)0, acc[mt][nt], false, false);
        }
        __syncthreads();
    }

#pragma unroll
    for (int mt = 0; mt < 2; ++mt)
#pragma unroll
        for (int nt = 0; nt < 4; ++nt)
#pragma unroll
            for (int v = 0; v < 8; ++v) {
                const int row = blockM + wm + mt * 16 + half * 8 + v;
                const int col = blockN + wn + nt * 16 + mrow;
                Y[(size_t)row * N_DIM + col] = acc[mt][nt][v];
            }
}

// ---------------------------------------------------------------------------
extern "C" void kernel_launch(void* const* d_in, const int* in_sizes, int n_in,
                              void* d_out, int out_size, void* d_ws, size_t ws_size,
                              hipStream_t stream) {
    const float* x     = (const float*)d_in[0];
    const int*   Wint  = (const int*)d_in[1];
    const float* lA    = (const float*)d_in[2];
    const float* lB    = (const float*)d_in[3];
    const float* scale = (const float*)d_in[4];
    const float* zp    = (const float*)d_in[5];
    float* out = (float*)d_out;

    const int M = in_sizes[0] / K_DIM;            // 8192

    unsigned short* Wb = (unsigned short*)d_ws;                 // 32 MB
    const size_t wbytes = (size_t)N_DIM * K_DIM * sizeof(unsigned short);
    const size_t xbytes = (size_t)M * K_DIM * sizeof(unsigned short);

    // 1) fused dequant + LoRA -> bf16 weight (fits in 192MB L2 afterwards)
    loftq_prep_w<<<(N_DIM * (K_DIM / 8)) / 256, 256, 0, stream>>>(Wint, lA, lB, scale, zp, Wb);

    if (ws_size >= wbytes + xbytes) {
        // 2) x -> bf16 once; GEMM moves tiles via async global->LDS DMA
        unsigned short* Xb = (unsigned short*)((char*)d_ws + wbytes);
        loftq_prep_x<<<((size_t)M * K_DIM / 8) / 256, 256, 0, stream>>>(x, Xb);
        dim3 grid(N_DIM / BN, M / BM);
        loftq_gemm_async<<<grid, 256, 0, stream>>>(Xb, Wb, out);
    } else {
        dim3 grid(N_DIM / 128, M / 128);
        loftq_gemm_f32x<<<grid, 256, 0, stream>>>(x, Wb, out);
    }
}
